// GAT_layer_53807350284465
// MI455X (gfx1250) — matrix operand order repaired and checked by reading, compiled, not run on hardware
//
#include <hip/hip_runtime.h>
#include <hip/hip_bf16.h>

// ---------------------------------------------------------------------------
// 2-layer GAT for MI455X (gfx1250, wave32).
// Dense projections: v_wmma_f32_16x16x32_bf16 (f32 accumulate), fragments
// loaded as wide aligned LDS vectors (ds_load_b128) from a transposed W tile.
// Edge softmax/aggregation: L2-resident float atomics (<192MB working set).
// ---------------------------------------------------------------------------

typedef __attribute__((ext_vector_type(16))) __bf16 v16bf;
typedef __attribute__((ext_vector_type(8)))  __bf16 v8bf;
typedef __attribute__((ext_vector_type(4)))  __bf16 v4bf;
typedef __attribute__((ext_vector_type(8)))  float  v8f;

#define NEG_SLOPE 0.2f
#define KDIM   128          // F_IN == H*HID == 128 (contraction dim, both GEMMs)
#define FOUT2  16
#define NGRAPH 64

// Monotone float -> uint mapping so unsigned atomicMax == float max.
// memset(0) is a valid "-infinity" initializer (every real float maps > 0).
__device__ __forceinline__ unsigned f2ord(float f) {
  unsigned u = __float_as_uint(f);
  return (u & 0x80000000u) ? ~u : (u | 0x80000000u);
}
__device__ __forceinline__ float ord2f(unsigned k) {
  return __uint_as_float((k & 0x80000000u) ? (k & 0x7fffffffu) : ~k);
}
__device__ __forceinline__ float lrelu(float v) {
  return v >= 0.0f ? v : NEG_SLOPE * v;
}

// ---------------------------------------------------------------------------
// Tiled GEMM: C[nrows x FOUT] = A[nrows x 128] * W[128 x FOUT]
// Block = 128 threads = 4 wave32; wave w owns rows [64*blk + 16w, +16).
//
// Fragment layouts (CDNA5 ISA §7.12.2, wave32):
//   A 16x32 bf16: lane l -> M = l&15; lane-half holds K 0..15 vs 16..31
//                 (same order as B, not ISA table order); VGPR r packs K pair (2r,2r+1) within chunk.
//   B 32x16 bf16: lane l -> N = l&15; this code's lane-half K 0..15 vs 16..31 (not the ISA table order),
//                 ascending K packed 2/VGPR -> 16 *contiguous* K per lane.
//   C 16x16 f32 : VGPR v, lane l -> M = v + (l<16?0:8), N = l&15.
// B is therefore one aligned v16bf LDS load from W^T; A is two v8bf loads.
// ---------------------------------------------------------------------------
#define XPAD 8    // sX row stride 136*2B = 272B  (multiple of 16, bank spread)
#define WPAD 16   // sWT row stride 144*2B = 288B (multiple of 32, bank spread)

template <int FOUT>
__global__ __launch_bounds__(128) void gemm_bf16_wmma(
    const float* __restrict__ A, const float* __restrict__ W,
    float* __restrict__ C, int nrows) {
  constexpr int NT = FOUT / 16;
  __shared__ __attribute__((aligned(32))) __bf16 sWT[FOUT][KDIM + WPAD]; // W^T
  __shared__ __attribute__((aligned(32))) __bf16 sX[64][KDIM + XPAD];
  const int tid = threadIdx.x;

  // Stage W transposed: sWT[n][k] = W[k*FOUT + n]. Coalesced global reads.
  for (int i = tid; i < KDIM * FOUT; i += 128) {
    const int k = i / FOUT, n = i % FOUT;
    sWT[n][k] = (__bf16)W[i];
  }

  // Stage 64 rows of A as bf16: float4 global loads -> packed b64 LDS stores.
  const int row0 = blockIdx.x * 64;
  for (int i = tid; i < 64 * (KDIM / 4); i += 128) {
    const int r = i >> 5, c4 = i & 31;           // KDIM/4 == 32
    const int gr = row0 + r;
    v4bf pk;
    if (gr < nrows) {
      const float4 f = *(const float4*)(A + (size_t)gr * KDIM + c4 * 4);
      pk[0] = (__bf16)f.x; pk[1] = (__bf16)f.y;
      pk[2] = (__bf16)f.z; pk[3] = (__bf16)f.w;
    } else {
      pk = (v4bf){(__bf16)0.f, (__bf16)0.f, (__bf16)0.f, (__bf16)0.f};
    }
    *(v4bf*)&sX[r][c4 * 4] = pk;
  }
  __syncthreads();

  const int wave = tid >> 5;
  const int lane = tid & 31;
  const int m    = lane & 15;
  const int half = lane >> 4;          // 0: low lane half, 1: high lane half
  const int arow = wave * 16 + m;

  v8f acc[NT] = {};

#pragma unroll
  for (int kt = 0; kt < KDIM / 32; ++kt) {
    const int K0 = kt * 32;
    // ---- A fragment: two contiguous 16B chunks, merged ----
    const int ka  = K0 + half * 16;
    const v8bf alo = *(const v8bf*)&sX[arow][ka];
    const v8bf ahi = *(const v8bf*)&sX[arow][ka + 8];
    const v16bf a = __builtin_shufflevector(alo, ahi, 0, 1, 2, 3, 4, 5, 6, 7,
                                            8, 9, 10, 11, 12, 13, 14, 15);
    const int kb = K0 + half * 16;
#pragma unroll
    for (int ct = 0; ct < NT; ++ct) {
      // ---- B fragment: 16 contiguous K at fixed column -> one v16bf ----
      const v16bf b = *(const v16bf*)&sWT[ct * 16 + m][kb];
      acc[ct] = __builtin_amdgcn_wmma_f32_16x16x32_bf16(
          false, a, false, b, (short)0, acc[ct], false, false);
    }
  }

#pragma unroll
  for (int ct = 0; ct < NT; ++ct) {
#pragma unroll
    for (int v = 0; v < 8; ++v) {
      const int grow = row0 + wave * 16 + half * 8 + v;
      if (grow < nrows) C[(size_t)grow * FOUT + ct * 16 + m] = acc[ct][v];
    }
  }
}

// ---------------------------------------------------------------------------
// Per-(node, head) attention coefficients: a_s = <h, att_src>, a_d = <h, att_dst>
// ---------------------------------------------------------------------------
template <int HEADS, int CHANS>
__global__ void attn_coef(const float* __restrict__ h,
                          const float* __restrict__ wsrc,
                          const float* __restrict__ wdst,
                          float* __restrict__ as_, float* __restrict__ ad_,
                          int n) {
  const int t = blockIdx.x * blockDim.x + threadIdx.x;
  if (t >= n * HEADS) return;
  const int node = t / HEADS, hh = t % HEADS;
  const float* hp = h + (size_t)node * (HEADS * CHANS) + hh * CHANS;
  float s = 0.f, d = 0.f;
#pragma unroll
  for (int c = 0; c < CHANS; ++c) {
    const float v = hp[c];
    s += v * wsrc[hh * CHANS + c];
    d += v * wdst[hh * CHANS + c];
  }
  as_[t] = s;
  ad_[t] = d;
}

// ---------------------------------------------------------------------------
// Edge pass 1: segment-max of leaky_relu(a_s[src] + a_d[dst]) per dst/head.
// ---------------------------------------------------------------------------
template <int HEADS>
__global__ void edge_max_k(const int* __restrict__ ei,
                           const float* __restrict__ as_,
                           const float* __restrict__ ad_,
                           unsigned* __restrict__ umax, int E) {
  const int t = blockIdx.x * blockDim.x + threadIdx.x;
  if (t >= E * HEADS) return;
  const int e = t / HEADS, hh = t % HEADS;
  const int s = ei[e], d = ei[E + e];
  const float v = lrelu(as_[s * HEADS + hh] + ad_[d * HEADS + hh]);
  atomicMax(&umax[d * HEADS + hh], f2ord(v));
}

// Edge pass 2: softmax denominator.
template <int HEADS>
__global__ void edge_expsum_k(const int* __restrict__ ei,
                              const float* __restrict__ as_,
                              const float* __restrict__ ad_,
                              const unsigned* __restrict__ umax,
                              float* __restrict__ den, int E) {
  const int t = blockIdx.x * blockDim.x + threadIdx.x;
  if (t >= E * HEADS) return;
  const int e = t / HEADS, hh = t % HEADS;
  const int s = ei[e], d = ei[E + e];
  const int idx = d * HEADS + hh;
  const float v = lrelu(as_[s * HEADS + hh] + ad_[idx]);
  atomicAdd(&den[idx], expf(v - ord2f(umax[idx])));
}

// Edge pass 3: out[dst] += alpha * h[src]  (alpha recomputed; avoids storing
// E x H floats). Prefetch the gathered row before the dependent scalar loads,
// then float4 gathers + f32 atomic scatter (all L2-resident).
template <int HEADS, int CHANS>
__global__ void edge_aggr_k(const int* __restrict__ ei,
                            const float* __restrict__ as_,
                            const float* __restrict__ ad_,
                            const unsigned* __restrict__ umax,
                            const float* __restrict__ den,
                            const float* __restrict__ h,
                            float* __restrict__ out, int E) {
  const int t = blockIdx.x * blockDim.x + threadIdx.x;
  if (t >= E * HEADS) return;
  const int e = t / HEADS, hh = t % HEADS;
  const int s = ei[e], d = ei[E + e];
  const float4* hp =
      (const float4*)(h + (size_t)s * (HEADS * CHANS) + hh * CHANS);
  __builtin_prefetch(hp, 0, 1);   // global_prefetch_b8: start the gather early
  const int idx = d * HEADS + hh;
  const float v = lrelu(as_[s * HEADS + hh] + ad_[idx]);
  const float alpha = expf(v - ord2f(umax[idx])) / (den[idx] + 1e-16f);
  float* op = out + (size_t)d * (HEADS * CHANS) + hh * CHANS;
#pragma unroll
  for (int q = 0; q < CHANS / 4; ++q) {
    const float4 hv = hp[q];
    atomicAdd(&op[4 * q + 0], hv.x * alpha);
    atomicAdd(&op[4 * q + 1], hv.y * alpha);
    atomicAdd(&op[4 * q + 2], hv.z * alpha);
    atomicAdd(&op[4 * q + 3], hv.w * alpha);
  }
}

// out = elu(out + b1), in place (b1 broadcast over the 128-wide feature dim).
__global__ void bias_elu_k(float* __restrict__ x, const float* __restrict__ b,
                           int total) {
  const int t = blockIdx.x * blockDim.x + threadIdx.x;
  if (t >= total) return;
  const float v = x[t] + b[t & (KDIM - 1)];
  x[t] = v > 0.f ? v : expm1f(v);
}

// Pool: pooled[batch[n]] += h[n] + b2; counts[batch[n]] += 1.
__global__ void pool_sum_k(const float* __restrict__ h,
                           const float* __restrict__ b2,
                           const int* __restrict__ batch,
                           float* __restrict__ pooled,
                           float* __restrict__ counts, int n) {
  const int t = blockIdx.x * blockDim.x + threadIdx.x;
  if (t >= n * FOUT2) return;
  const int node = t >> 4, c = t & 15;
  const int g = batch[node];
  atomicAdd(&pooled[g * FOUT2 + c], h[t] + b2[c]);
  if (c == 0) atomicAdd(&counts[g], 1.0f);
}

// Final: mean + log_softmax over 16 classes for 64 graphs.
__global__ void pooled_logsoftmax_k(const float* __restrict__ pooled,
                                    const float* __restrict__ counts,
                                    float* __restrict__ out) {
  const int g = threadIdx.x;
  if (g >= NGRAPH) return;
  const float inv = 1.0f / fmaxf(counts[g], 1.0f);
  float v[FOUT2];
  float mx = -3.0e38f;
#pragma unroll
  for (int c = 0; c < FOUT2; ++c) {
    v[c] = pooled[g * FOUT2 + c] * inv;
    mx = fmaxf(mx, v[c]);
  }
  float s = 0.f;
#pragma unroll
  for (int c = 0; c < FOUT2; ++c) s += expf(v[c] - mx);
  const float lse = mx + logf(s);
#pragma unroll
  for (int c = 0; c < FOUT2; ++c) out[g * FOUT2 + c] = v[c] - lse;
}

// ---------------------------------------------------------------------------
extern "C" void kernel_launch(void* const* d_in, const int* in_sizes, int n_in,
                              void* d_out, int out_size, void* d_ws,
                              size_t ws_size, hipStream_t stream) {
  (void)n_in; (void)out_size; (void)ws_size;
  const float* x    = (const float*)d_in[0];
  const int*   ei   = (const int*)d_in[1];
  const int*   bat  = (const int*)d_in[2];
  const float* W1   = (const float*)d_in[3];
  const float* a1s  = (const float*)d_in[4];
  const float* a1d  = (const float*)d_in[5];
  const float* b1   = (const float*)d_in[6];
  const float* W2   = (const float*)d_in[7];
  const float* a2sw = (const float*)d_in[8];
  const float* a2dw = (const float*)d_in[9];
  const float* b2   = (const float*)d_in[10];
  const int N = in_sizes[2];       // batch has one entry per node
  const int E = in_sizes[1] / 2;   // edge_index is [2, E]

  // Workspace layout (floats). Layer-2 scratch aliases h1, which is dead after
  // the layer-1 aggregation pass (stream ordering guarantees safety).
  float* ws   = (float*)d_ws;
  float* h1   = ws;                           // N*128
  float* out1 = h1 + (size_t)N * 128;         // N*128
  float* as1  = out1 + (size_t)N * 128;       // N*8
  float* ad1  = as1 + (size_t)N * 8;          // N*8
  unsigned* umax1 = (unsigned*)(ad1 + (size_t)N * 8);  // N*8
  float* den1 = (float*)umax1 + (size_t)N * 8;         // N*8
  float* pooled = den1 + (size_t)N * 8;       // 64*16
  float* counts = pooled + NGRAPH * FOUT2;    // 64
  // layer-2 aliases (inside old h1 region, N*128 >= N*36):
  float* h3   = h1;                           // N*16
  float* out2 = h1 + (size_t)N * 16;          // N*16
  float* as2  = h1 + (size_t)N * 32;          // N
  float* ad2  = as2 + N;                      // N
  unsigned* umax2 = (unsigned*)(ad2 + N);     // N
  float* den2 = (float*)umax2 + N;            // N (right after umax2)

  const int TB = 256;

  // ---------------- layer 1 ----------------
  hipMemsetAsync(out1, 0, (size_t)N * 128 * sizeof(float), stream);
  hipMemsetAsync(umax1, 0, (size_t)N * 8 * sizeof(float) * 2, stream); // +den1
  hipMemsetAsync(pooled, 0, (NGRAPH * FOUT2 + NGRAPH) * sizeof(float), stream);

  gemm_bf16_wmma<128>
      <<<dim3((N + 63) / 64), dim3(128), 0, stream>>>(x, W1, h1, N);
  attn_coef<8, 16>
      <<<dim3((N * 8 + TB - 1) / TB), dim3(TB), 0, stream>>>(h1, a1s, a1d,
                                                             as1, ad1, N);
  edge_max_k<8>
      <<<dim3((E * 8 + TB - 1) / TB), dim3(TB), 0, stream>>>(ei, as1, ad1,
                                                             umax1, E);
  edge_expsum_k<8>
      <<<dim3((E * 8 + TB - 1) / TB), dim3(TB), 0, stream>>>(ei, as1, ad1,
                                                             umax1, den1, E);
  edge_aggr_k<8, 16>
      <<<dim3((E * 8 + TB - 1) / TB), dim3(TB), 0, stream>>>(
          ei, as1, ad1, umax1, den1, h1, out1, E);
  bias_elu_k<<<dim3((N * 128 + TB - 1) / TB), dim3(TB), 0, stream>>>(
      out1, b1, N * 128);

  // ---------------- layer 2 ----------------
  hipMemsetAsync(out2, 0, (size_t)N * 16 * sizeof(float), stream);
  hipMemsetAsync(umax2, 0, (size_t)N * sizeof(float) * 2, stream);    // +den2

  gemm_bf16_wmma<16>
      <<<dim3((N + 63) / 64), dim3(128), 0, stream>>>(out1, W2, h3, N);
  attn_coef<1, 16>
      <<<dim3((N + TB - 1) / TB), dim3(TB), 0, stream>>>(h3, a2sw, a2dw,
                                                         as2, ad2, N);
  edge_max_k<1>
      <<<dim3((E + TB - 1) / TB), dim3(TB), 0, stream>>>(ei, as2, ad2,
                                                         umax2, E);
  edge_expsum_k<1>
      <<<dim3((E + TB - 1) / TB), dim3(TB), 0, stream>>>(ei, as2, ad2,
                                                         umax2, den2, E);
  edge_aggr_k<1, 16>
      <<<dim3((E + TB - 1) / TB), dim3(TB), 0, stream>>>(
          ei, as2, ad2, umax2, den2, h3, out2, E);

  // ---------------- readout ----------------
  pool_sum_k<<<dim3((N * 16 + TB - 1) / TB), dim3(TB), 0, stream>>>(
      out2, b2, bat, pooled, counts, N);
  pooled_logsoftmax_k<<<dim3(1), dim3(64), 0, stream>>>(pooled, counts,
                                                        (float*)d_out);
}
